// Attention_31842887532892
// MI455X (gfx1250) — matrix operand order repaired and checked by reading, not compiled, not run on hardware
//
#include <hip/hip_runtime.h>
#include <stddef.h>

typedef __attribute__((ext_vector_type(16))) __bf16 v16bf;
typedef __attribute__((ext_vector_type(8)))  __bf16 v8bf;
typedef __attribute__((ext_vector_type(8)))  float  v8f;

constexpr int Bsz = 2, T = 2048, C = 1024, H = 16, Dh = 64;

#define DEVFN static __device__ __forceinline__

// A-fragment (16x32 bf16): lane m=L%16, h=L>>4 holds K = h*16+0..7 (elems 0-7)
// and K = h*16+8..15 (elems 8-15). Two contiguous 16B chunks.
DEVFN v16bf afrag(const __bf16* p0, const __bf16* p1) {
    v8bf a = *(const v8bf*)p0;
    v8bf b = *(const v8bf*)p1;
    return __builtin_shufflevector(a, b, 0,1,2,3,4,5,6,7,8,9,10,11,12,13,14,15);
}
// B-fragment (32x16 bf16): lane n=L%16, h=L>>4 holds K = h*16+0..15, one 32B chunk.
DEVFN v16bf bfrag(const __bf16* p) { return *(const v16bf*)p; }

DEVFN v8f wmma_bf16(v16bf a, v16bf b, v8f c) {
    return __builtin_amdgcn_wmma_f32_16x16x32_bf16(false, a, false, b, (short)0, c,
                                                   false, false);
}
// swap values between lane L and L^16 (ds_swizzle SWAPX16: xor=0x10, and=0x1f)
DEVFN float swap16(float x) {
    return __int_as_float(__builtin_amdgcn_ds_swizzle(__float_as_int(x), 0x401F));
}

DEVFN v8f vzero() { return (v8f){0.f, 0.f, 0.f, 0.f, 0.f, 0.f, 0.f, 0.f}; }

// ---------------------------------------------------------------- f32 -> bf16
__global__ void cvt_bf16(const float* __restrict__ s, __bf16* __restrict__ d, int n) {
    int i = blockIdx.x * blockDim.x + threadIdx.x;
    if (i < n) d[i] = (__bf16)s[i];
}

// --------------------------------------------- Q/K projection: out[B,H,T,64]
// y[t,o] = sum_c x[t,c] * W[o,c];  A = x rows, B = W rows (W is [out,in]).
// Register-blocked: one wave computes a 32(t) x 32(o) tile -> 2 loads / WMMA.
__global__ __launch_bounds__(256) void proj_qk(const __bf16* __restrict__ xb,
                                               const __bf16* __restrict__ Wb,
                                               __bf16* __restrict__ out) {
    int wave = blockIdx.x * (blockDim.x >> 5) + (threadIdx.x >> 5);
    int lane = threadIdx.x & 31, n = lane & 15, hh = lane >> 4;
    int oo = wave % (C / 32);
    int rem = wave / (C / 32);
    int tt = rem % (T / 32);
    int b  = rem / (T / 32);
    int t0 = tt * 32, o0 = oo * 32;

    const __bf16* xr0 = xb + (size_t)(b * T + t0 + n) * C;       // A rows (m -> t)
    const __bf16* xr1 = xb + (size_t)(b * T + t0 + 16 + n) * C;
    const __bf16* wr0 = Wb + (size_t)(o0 + n) * C;               // B cols (n -> o)
    const __bf16* wr1 = Wb + (size_t)(o0 + 16 + n) * C;
    v8f acc[2][2] = {{vzero(), vzero()}, {vzero(), vzero()}};
#pragma unroll 4
    for (int kc = 0; kc < C; kc += 32) {
        v16bf a0 = afrag(xr0 + kc + hh * 16, xr0 + kc + 8 + hh * 16);
        v16bf a1 = afrag(xr1 + kc + hh * 16, xr1 + kc + 8 + hh * 16);
        v16bf b0 = bfrag(wr0 + kc + hh * 16);
        v16bf b1 = bfrag(wr1 + kc + hh * 16);
        acc[0][0] = wmma_bf16(a0, b0, acc[0][0]);
        acc[0][1] = wmma_bf16(a0, b1, acc[0][1]);
        acc[1][0] = wmma_bf16(a1, b0, acc[1][0]);
        acc[1][1] = wmma_bf16(a1, b1, acc[1][1]);
    }
    // D layout: lane n -> col o ; VGPR r -> row t = +r + 8*hh
#pragma unroll
    for (int ni = 0; ni < 2; ni++) {
        int o = o0 + ni * 16 + n, hd = o >> 6, d = o & 63;
        size_t base = (size_t)(b * H + hd) * T * Dh + d;
#pragma unroll
        for (int mi = 0; mi < 2; mi++)
#pragma unroll
            for (int r = 0; r < 8; r++) {
                int t = t0 + mi * 16 + r + 8 * hh;
                out[base + (size_t)t * Dh] = (__bf16)acc[mi][ni][r];
            }
    }
}

// ----------------------------------- V projection, transposed: vt[B,H,64,T]
// Roles swapped: A = W rows (m -> o), B = x rows (n -> t) => D rows are o.
__global__ __launch_bounds__(256) void proj_v(const __bf16* __restrict__ xb,
                                              const __bf16* __restrict__ Wb,
                                              __bf16* __restrict__ vt) {
    int wave = blockIdx.x * (blockDim.x >> 5) + (threadIdx.x >> 5);
    int lane = threadIdx.x & 31, n = lane & 15, hh = lane >> 4;
    int oo = wave % (C / 32);
    int rem = wave / (C / 32);
    int tt = rem % (T / 32);
    int b  = rem / (T / 32);
    int t0 = tt * 32, o0 = oo * 32;

    const __bf16* wr0 = Wb + (size_t)(o0 + n) * C;               // A rows (m -> o)
    const __bf16* wr1 = Wb + (size_t)(o0 + 16 + n) * C;
    const __bf16* xr0 = xb + (size_t)(b * T + t0 + n) * C;       // B cols (n -> t)
    const __bf16* xr1 = xb + (size_t)(b * T + t0 + 16 + n) * C;
    v8f acc[2][2] = {{vzero(), vzero()}, {vzero(), vzero()}};
#pragma unroll 4
    for (int kc = 0; kc < C; kc += 32) {
        v16bf a0 = afrag(wr0 + kc + hh * 16, wr0 + kc + 8 + hh * 16);
        v16bf a1 = afrag(wr1 + kc + hh * 16, wr1 + kc + 8 + hh * 16);
        v16bf b0 = bfrag(xr0 + kc + hh * 16);
        v16bf b1 = bfrag(xr1 + kc + hh * 16);
        acc[0][0] = wmma_bf16(a0, b0, acc[0][0]);
        acc[0][1] = wmma_bf16(a0, b1, acc[0][1]);
        acc[1][0] = wmma_bf16(a1, b0, acc[1][0]);
        acc[1][1] = wmma_bf16(a1, b1, acc[1][1]);
    }
    // lane n -> t ; VGPR r -> o = +r + 8*hh
#pragma unroll
    for (int ni = 0; ni < 2; ni++) {
        int t = t0 + ni * 16 + n;
#pragma unroll
        for (int mi = 0; mi < 2; mi++)
#pragma unroll
            for (int r = 0; r < 8; r++) {
                int o = o0 + mi * 16 + r + 8 * hh, hd = o >> 6, d = o & 63;
                vt[((size_t)(b * H + hd) * Dh + d) * T + t] = (__bf16)acc[mi][ni][r];
            }
    }
}

// -------------------------------------------------- flash attention (per wave:
// one (b,h) head, one 16-query tile, streamed over 32-key chunks, causal)
__global__ __launch_bounds__(256) void attn(const __bf16* __restrict__ Q,
                                            const __bf16* __restrict__ Kb,
                                            const __bf16* __restrict__ Vt,
                                            __bf16* __restrict__ Y) {
    __shared__ __align__(64) __bf16 Pb[8][16 * 48];  // per-wave P staging, stride 48
    int wv   = threadIdx.x >> 5;
    int wave = blockIdx.x * 8 + wv;
    int lane = threadIdx.x & 31, n = lane & 15, hh = lane >> 4;
    int qt = wave % (T / 16);
    int rem = wave / (T / 16);
    int h = rem % H, b = rem / H;
    int q0 = qt * 16;

    const __bf16* qrow  = Q  + ((size_t)(b * H + h) * T + q0 + n) * Dh;
    const __bf16* kbase = Kb + (size_t)(b * H + h) * T * Dh;
    const __bf16* vbase = Vt + (size_t)(b * H + h) * Dh * T;

    // Q as B-operand of S^T = K * Q^T : lane n -> query col, split over d.
    v16bf bq0 = bfrag(qrow + hh * 16);        // d = 0..31
    v16bf bq1 = bfrag(qrow + 32 + hh * 16);   // d = 32..63

    v8f ov[4];
#pragma unroll
    for (int dt = 0; dt < 4; dt++) ov[dt] = vzero();
    float m = -3.0e38f, l = 0.f;
    const float scale = 0.125f;  // 1/sqrt(64)

    __bf16* prow = &Pb[wv][n * 48];
    int nch = qt / 2 + 1;  // chunks of 32 keys with k0 <= q0+15
    for (int c = 0; c < nch; c++) {
        int k0 = c * 32;
        // S^T tiles: M=16 keys, N=16 queries, K=d=64 (2 WMMAs each)
        v8f cs[2];
#pragma unroll
        for (int sub = 0; sub < 2; sub++) {
            const __bf16* krow = kbase + (size_t)(k0 + sub * 16 + n) * Dh;
            v8f s8 = vzero();
            s8 = wmma_bf16(afrag(krow + hh * 16, krow + 8 + hh * 16), bq0, s8);
            s8 = wmma_bf16(afrag(krow + 32 + hh * 16, krow + 40 + hh * 16), bq1, s8);
            cs[sub] = s8;
        }
        // per-lane: 16 scores for query q0+n, keys k0 + sub*16 + 8*hh + r
        float sv[16];
#pragma unroll
        for (int sub = 0; sub < 2; sub++)
#pragma unroll
            for (int r = 0; r < 8; r++) {
                int key  = k0 + sub * 16 + 8 * hh + r;
                float v  = cs[sub][r] * scale;
                sv[sub * 8 + r] = (key > q0 + n) ? -3.0e38f : v;
            }
        float mx = sv[0];
#pragma unroll
        for (int i = 1; i < 16; i++) mx = fmaxf(mx, sv[i]);
        mx = fmaxf(mx, swap16(mx));          // combine lane-pair (other key half)
        float mn = fmaxf(m, mx);
        float alpha = __expf(m - mn);
        m = mn;
        float ps = 0.f;
#pragma unroll
        for (int i = 0; i < 16; i++) { sv[i] = __expf(sv[i] - mn); ps += sv[i]; }
        ps += swap16(ps);
        l = l * alpha + ps;
#pragma unroll
        for (int dt = 0; dt < 4; dt++)
#pragma unroll
            for (int r = 0; r < 8; r++) ov[dt][r] *= alpha;

        // stage P (bf16) to LDS: row = query n, cols = key-in-chunk
#pragma unroll
        for (int sub = 0; sub < 2; sub++) {
            v8bf pk;
#pragma unroll
            for (int r = 0; r < 8; r++) pk[r] = (__bf16)sv[sub * 8 + r];
            *(v8bf*)(prow + sub * 16 + hh * 8) = pk;   // keys sub*16 + 8*hh ..+7
        }
        asm volatile("s_wait_dscnt 0" ::: "memory");   // same-wave LDS RAW
        // reload as B-fragment of P^T (K=32 keys, N=16 queries)
        v16bf pbf = *(const v16bf*)(&Pb[wv][n * 48 + hh * 16]);
        // O^T += V^T * P^T : A rows = d (from Vt, contiguous keys)
#pragma unroll
        for (int dt = 0; dt < 4; dt++) {
            const __bf16* vrow = vbase + (size_t)(dt * 16 + n) * T + k0;
            v16bf va = afrag(vrow + hh * 16, vrow + 8 + hh * 16);
            ov[dt] = wmma_bf16(va, pbf, ov[dt]);
        }
    }
    float rl = 1.f / l;
    __bf16* yb = Y + (size_t)(b * T + q0 + n) * C + h * Dh;
#pragma unroll
    for (int dt = 0; dt < 4; dt++)
#pragma unroll
        for (int r = 0; r < 8; r++) {
            int d = dt * 16 + r + 8 * hh;   // O^T: lane n -> query, VGPR r -> d
            yb[d] = (__bf16)(ov[dt][r] * rl);
        }
}

// ------------------------------------------- output projection, f32 to d_out
// Register-blocked 32(t) x 32(o) like proj_qk.
__global__ __launch_bounds__(256) void out_gemm(const __bf16* __restrict__ Yb,
                                                const __bf16* __restrict__ Wb,
                                                float* __restrict__ out) {
    int wave = blockIdx.x * (blockDim.x >> 5) + (threadIdx.x >> 5);
    int lane = threadIdx.x & 31, n = lane & 15, hh = lane >> 4;
    int oo = wave % (C / 32);
    int rem = wave / (C / 32);
    int tt = rem % (T / 32);
    int b  = rem / (T / 32);
    int t0 = tt * 32, o0 = oo * 32;

    const __bf16* yr0 = Yb + (size_t)(b * T + t0 + n) * C;
    const __bf16* yr1 = Yb + (size_t)(b * T + t0 + 16 + n) * C;
    const __bf16* wr0 = Wb + (size_t)(o0 + n) * C;
    const __bf16* wr1 = Wb + (size_t)(o0 + 16 + n) * C;
    v8f acc[2][2] = {{vzero(), vzero()}, {vzero(), vzero()}};
#pragma unroll 4
    for (int kc = 0; kc < C; kc += 32) {
        v16bf a0 = afrag(yr0 + kc + hh * 16, yr0 + kc + 8 + hh * 16);
        v16bf a1 = afrag(yr1 + kc + hh * 16, yr1 + kc + 8 + hh * 16);
        v16bf b0 = bfrag(wr0 + kc + hh * 16);
        v16bf b1 = bfrag(wr1 + kc + hh * 16);
        acc[0][0] = wmma_bf16(a0, b0, acc[0][0]);
        acc[0][1] = wmma_bf16(a0, b1, acc[0][1]);
        acc[1][0] = wmma_bf16(a1, b0, acc[1][0]);
        acc[1][1] = wmma_bf16(a1, b1, acc[1][1]);
    }
#pragma unroll
    for (int mi = 0; mi < 2; mi++)
#pragma unroll
        for (int r = 0; r < 8; r++) {
            int t = t0 + mi * 16 + r + 8 * hh;
            float* orow = out + (size_t)(b * T + t) * C + o0 + n;
            orow[0]  = acc[mi][0][r];   // coalesced f32 stores
            orow[16] = acc[mi][1][r];
        }
}

extern "C" void kernel_launch(void* const* d_in, const int* in_sizes, int n_in,
                              void* d_out, int out_size, void* d_ws, size_t ws_size,
                              hipStream_t stream) {
    (void)in_sizes; (void)n_in; (void)out_size; (void)ws_size;
    const float* x  = (const float*)d_in[0];
    const float* Wq = (const float*)d_in[1];
    const float* Wk = (const float*)d_in[2];
    const float* Wv = (const float*)d_in[3];
    const float* Wo = (const float*)d_in[4];
    float* out = (float*)d_out;

    const size_t NXT = (size_t)Bsz * T * C;  // 4M
    const size_t NW  = (size_t)C * C;        // 1M
    char* w = (char*)d_ws;
    __bf16* xb  = (__bf16*)w;  w += NXT * 2;
    __bf16* wqb = (__bf16*)w;  w += NW * 2;
    __bf16* wkb = (__bf16*)w;  w += NW * 2;
    __bf16* wvb = (__bf16*)w;  w += NW * 2;
    __bf16* wob = (__bf16*)w;  w += NW * 2;
    __bf16* Qb  = (__bf16*)w;  w += NXT * 2;  // [B,H,T,64]
    __bf16* Kb  = (__bf16*)w;  w += NXT * 2;  // [B,H,T,64]
    __bf16* Vt  = (__bf16*)w;  w += NXT * 2;  // [B,H,64,T]
    __bf16* Yb  = (__bf16*)w;  w += NXT * 2;  // [B,T,C]

    cvt_bf16<<<(int)((NXT + 255) / 256), 256, 0, stream>>>(x,  xb,  (int)NXT);
    cvt_bf16<<<(int)((NW  + 255) / 256), 256, 0, stream>>>(Wq, wqb, (int)NW);
    cvt_bf16<<<(int)((NW  + 255) / 256), 256, 0, stream>>>(Wk, wkb, (int)NW);
    cvt_bf16<<<(int)((NW  + 255) / 256), 256, 0, stream>>>(Wv, wvb, (int)NW);
    cvt_bf16<<<(int)((NW  + 255) / 256), 256, 0, stream>>>(Wo, wob, (int)NW);

    // GEMM tile waves: B * (T/32) * (C/32) = 4096 ; 8 waves / 256-thread block
    proj_qk<<<512, 256, 0, stream>>>(xb, wqb, Qb);
    proj_qk<<<512, 256, 0, stream>>>(xb, wkb, Kb);
    proj_v <<<512, 256, 0, stream>>>(xb, wvb, Vt);

    // attention waves: B*H*(T/16) = 4096 ; 8 waves / block
    attn<<<512, 256, 0, stream>>>(Qb, Kb, Vt, Yb);

    out_gemm<<<512, 256, 0, stream>>>(Yb, wob, out);
}
